// LTC1_54949811585272
// MI455X (gfx1250) — matrix operand order repaired and checked by reading, not compiled, not run on hardware
//
#include <hip/hip_runtime.h>

typedef __attribute__((ext_vector_type(16))) _Float16 v16h;
typedef __attribute__((ext_vector_type(8)))  float    v8f;

#define BATCH   8192
#define SEQ     32
#define ISZ     16
#define USZ     19
#define UNFOLDS 6
#define EPSV    1e-8f

// Weighted sigmoid: returns wh*(1 + tanh(arg)) == (2*wh) * sigmoid(2*arg).
// With wh = 0.5*w and arg = 0.5*sigma*(v-mu) this equals w*sigmoid(sigma*(v-mu))
// at ONE transcendental op (v_tanh_f32) instead of exp+rcp.
__device__ __forceinline__ float wsig(float wh, float arg) {
#if __has_builtin(__builtin_amdgcn_tanhf)
  return fmaf(wh, __builtin_amdgcn_tanhf(arg), wh);
#else
  // sigmoid(2*arg) = 1/(1+exp2(-2*log2(e)*arg))
  return (wh * 2.f) *
         __builtin_amdgcn_rcpf(1.f + __builtin_amdgcn_exp2f(arg * -2.885390081777927f));
#endif
}

// -------------------------------------------------------------------------
// Kernel 1: conv1 (1->6, 5x5 valid) + bias + relu + 2x2 maxpool
//   in : x (B,1,28,44) f32
//   out: pooled1 (B,6,12,20) stored f16 (feeds the WMMA GEMM)
// -------------------------------------------------------------------------
__global__ __launch_bounds__(256) void conv1_pool_kernel(
    const float* __restrict__ x, const float* __restrict__ w1,
    const float* __restrict__ b1, _Float16* __restrict__ pooled1)
{
  __shared__ float ximg[28 * 44];
  __shared__ float ws[150];
  __shared__ float bs[6];
  const int b = blockIdx.x;
  const int tid = threadIdx.x;
  for (int i = tid; i < 28 * 44; i += 256) ximg[i] = x[b * 28 * 44 + i];
  for (int i = tid; i < 150; i += 256) ws[i] = w1[i];
  if (tid < 6) bs[tid] = b1[tid];
  __syncthreads();

  for (int c = 0; c < 6; ++c) {
    for (int o = tid; o < 240; o += 256) {        // 12 x 20 pooled outputs
      const int ph = o / 20, pw = o % 20;
      float m = -3.4e38f;
      #pragma unroll
      for (int dy = 0; dy < 2; ++dy) {
        #pragma unroll
        for (int dx = 0; dx < 2; ++dx) {
          float acc = 0.f;
          #pragma unroll
          for (int kh = 0; kh < 5; ++kh)
            #pragma unroll
            for (int kw = 0; kw < 5; ++kw)
              acc = fmaf(ximg[(2 * ph + dy + kh) * 44 + (2 * pw + dx + kw)],
                         ws[c * 25 + kh * 5 + kw], acc);
          m = fmaxf(m, acc);
        }
      }
      const float r = fmaxf(m + bs[c], 0.f);      // relu(max+bias) == pool(relu(conv+bias))
      pooled1[((b * 6 + c) * 12 + ph) * 20 + pw] = (_Float16)r;
    }
  }
}

// -------------------------------------------------------------------------
// Kernel 2: conv2 (6->16, 5x5 valid) as WMMA implicit GEMM + bias + relu + pool
//   GEMM per image: M=128 output pixels (8x16), N=16 oc.
//   K re-tiled as 6 steps x 32 (one input channel per step, taps 0..24 valid,
//   25..31 zero-padded) so every fragment K-offset is a compile-time constant
//   (two-way select on lane-half) -- no runtime div/mod in the hot loop.
//   One workgroup per image; 8 waves, wave w owns conv-output row oh=w.
//   out: feat (B, 512) f32 flat = oc*32 + ph*8 + pw  (matches reference reshape)
// -------------------------------------------------------------------------
__global__ __launch_bounds__(256) void conv2_wmma_kernel(
    const _Float16* __restrict__ pooled1, const float* __restrict__ w2,
    const float* __restrict__ b2, float* __restrict__ feat)
{
  __shared__ __align__(32) _Float16 wfrag[6 * 32 * 16]; // B fragments, lane-major
  __shared__ __align__(4)  _Float16 tile[6 * 12 * 20];  // activation tile (1440 f16)
  __shared__ float olds[128 * 16];                      // conv2 raw output (pixel x oc)
  const int b = blockIdx.x;
  const int tid = threadIdx.x;
  const int lane = tid & 31;
  const int wave = tid >> 5;            // 0..7 == conv2 output row oh

  // Stage B in fragment order: wfrag[(t*32+l)*16 + e] = W[k = t*25 + tap][n=l&15]
  // with tap = ((l>>4)<<3) + e + ((e>>3)<<3); zero-pad taps 25..31. Each lane later reads its
  // whole fragment as one contiguous 32-byte v16h (ds_load_b128 pair).
  for (int i = tid; i < 6 * 32 * 16; i += 256) {
    const int e = i & 15, l = (i >> 4) & 31, t = i >> 9;
    const int n = l & 15, tap = ((l >> 4) << 3) + e + ((e >> 3) << 3);
    wfrag[i] = (tap < 25) ? (_Float16)w2[n * 150 + t * 25 + tap] : (_Float16)0.f;
  }
  // Stage this image's pooled1 tile (coalesced 32-bit copies; 2880 B)
  {
    const unsigned int* src = (const unsigned int*)(pooled1 + (size_t)b * 1440);
    unsigned int* dst = (unsigned int*)tile;
    for (int i = tid; i < 720; i += 256) dst[i] = src[i];
  }
  __syncthreads();

  const int oh = wave;
  const int ow = lane & 15;             // this lane's A-row pixel column
  const int hi = (lane >> 4) & 1;
  v8f acc = {0.f, 0.f, 0.f, 0.f, 0.f, 0.f, 0.f, 0.f};

  #pragma unroll
  for (int t = 0; t < 6; ++t) {         // t == input channel
    const v16h bfrag = *(const v16h*)&wfrag[(t * 32 + lane) * 16];
    const int abase = (t * 12 + oh) * 20 + ow;
    v16h afrag = {};
    #pragma unroll
    for (int v = 0; v < 8; ++v) {
      #pragma unroll
      for (int h = 0; h < 2; ++h) {
        // ISA 7.12.2 16-bit A layout: lane<16 tap=c0, lane>=16 tap=c0+8
        const int c0   = ((v & 3) << 1) + h + ((v >> 2) << 4);   // 0..7,16..23
        const int tap1 = c0 + 8;                                 // 8..15,24..31
        const int off0 = (c0 / 5) * 20 + (c0 % 5);               // compile-time
        const int off1 = (tap1 < 25) ? (tap1 / 5) * 20 + (tap1 % 5) : off0;
        _Float16 val = tile[abase + (hi ? off1 : off0)];
        if (tap1 >= 25) val = hi ? (_Float16)0.f : val;          // pad region
        afrag[v * 2 + h] = val;
      }
    }
    acc = __builtin_amdgcn_wmma_f32_16x16x32_f16(false, afrag, false, bfrag,
                                                 (short)0, acc, false, false);
  }

  // C layout: VGPR i -> row M = i + 8*hi, col N = lane&15 ; pixel p = wave*16 + M
  {
    const int n = lane & 15;
    const int pbase = wave * 16 + hi * 8;
    #pragma unroll
    for (int i = 0; i < 8; ++i) olds[(pbase + i) * 16 + n] = acc[i];
  }
  __syncthreads();

  // bias + relu + 2x2 maxpool -> feat[b, oc*32 + ph*8 + pw]
  for (int o = tid; o < 512; o += 256) {
    const int oc = o >> 5, rem = o & 31;
    const int ph = rem >> 3, pw = rem & 7;
    const int p0 = (2 * ph) * 16 + 2 * pw;        // pixel index = oh*16 + ow
    const float v0 = olds[p0 * 16 + oc];
    const float v1 = olds[(p0 + 1) * 16 + oc];
    const float v2 = olds[(p0 + 16) * 16 + oc];
    const float v3 = olds[(p0 + 17) * 16 + oc];
    const float m = fmaxf(fmaxf(v0, v1), fmaxf(v2, v3));
    feat[b * 512 + o] = fmaxf(m + b2[oc], 0.f);
  }
}

// -------------------------------------------------------------------------
// Kernel 3: LTC recurrence, one batch element per lane, params in LDS.
//   wa[b,j,u] = w[j,u]*sigmoid(sig[j,u]*(v[b,j]-mu[j,u])) is a per-(j,u)
//   nonlinearity of v -> transcendental-bound. Prefold constants so each
//   synapse is: fma (arg) + v_tanh + fma (weight) + fma (num) + add (den).
//   LDS tables: Lh = 0.5*sigma, Lnm = -0.5*sigma*mu, Lwh = 0.5*w*mask, Le = erev.
// -------------------------------------------------------------------------
__global__ __launch_bounds__(256) void ltc_kernel(
    const float* __restrict__ feat,
    const float* __restrict__ gleak, const float* __restrict__ vleak,
    const float* __restrict__ cm,
    const float* __restrict__ sigma, const float* __restrict__ mu,
    const float* __restrict__ w, const float* __restrict__ erev,
    const float* __restrict__ ssig, const float* __restrict__ smu,
    const float* __restrict__ sw, const float* __restrict__ serev,
    const float* __restrict__ iw, const float* __restrict__ ib,
    const float* __restrict__ ow_, const float* __restrict__ ob,
    const float* __restrict__ smask, const float* __restrict__ ssmask,
    float* __restrict__ out)
{
  __shared__ float Lh[USZ * USZ], Lnm[USZ * USZ], Lwh[USZ * USZ], Le[USZ * USZ];
  __shared__ float Sh[ISZ * USZ], Snm[ISZ * USZ], Swh[ISZ * USZ], Se[ISZ * USZ];
  __shared__ float Lg[USZ], Lgv[USZ], Lct[USZ];
  const int tid = threadIdx.x;
  for (int i = tid; i < USZ * USZ; i += 256) {
    const float sg = 0.5f * sigma[i];
    Lh[i]  = sg;
    Lnm[i] = -sg * mu[i];
    Lwh[i] = 0.5f * w[i] * smask[i];    // fold sparsity mask + 1/2 into weight
    Le[i]  = erev[i];
  }
  for (int i = tid; i < ISZ * USZ; i += 256) {
    const float sg = 0.5f * ssig[i];
    Sh[i]  = sg;
    Snm[i] = -sg * smu[i];
    Swh[i] = 0.5f * sw[i] * ssmask[i];
    Se[i]  = serev[i];
  }
  if (tid < USZ) {
    Lg[tid]  = gleak[tid];
    Lgv[tid] = gleak[tid] * vleak[tid];
    Lct[tid] = cm[tid] * (float)UNFOLDS;   // cm / (dt/unfolds), dt = 1
  }
  __syncthreads();

  const int b = blockIdx.x * 256 + tid;

  float v[USZ];
  #pragma unroll
  for (int u = 0; u < USZ; ++u) v[u] = 0.f;

  #pragma unroll 1
  for (int s = 0; s < SEQ; ++s) {
    float wnum[USZ], wden[USZ];
    #pragma unroll
    for (int u = 0; u < USZ; ++u) { wnum[u] = 0.f; wden[u] = 0.f; }
    #pragma unroll
    for (int i = 0; i < ISZ; ++i) {
      const float xi = feat[b * 512 + s * 16 + i] * iw[i] + ib[i];
      #pragma unroll
      for (int u = 0; u < USZ; ++u) {
        const int idx = i * USZ + u;
        const float a = wsig(Swh[idx], fmaf(Sh[idx], xi, Snm[idx]));
        wnum[u] = fmaf(a, Se[idx], wnum[u]);
        wden[u] += a;
      }
    }
    #pragma unroll 1
    for (int f = 0; f < UNFOLDS; ++f) {
      float num[USZ], den[USZ];
      #pragma unroll
      for (int u = 0; u < USZ; ++u) {
        num[u] = Lct[u] * v[u] + Lgv[u] + wnum[u];
        den[u] = Lct[u] + Lg[u] + wden[u];
      }
      #pragma unroll
      for (int j = 0; j < USZ; ++j) {
        const float vj = v[j];
        #pragma unroll
        for (int u = 0; u < USZ; ++u) {
          const int idx = j * USZ + u;
          const float a = wsig(Lwh[idx], fmaf(Lh[idx], vj, Lnm[idx]));
          num[u] = fmaf(a, Le[idx], num[u]);
          den[u] += a;
        }
      }
      #pragma unroll
      for (int u = 0; u < USZ; ++u)
        v[u] = num[u] * __builtin_amdgcn_rcpf(den[u] + EPSV);
    }
  }
  out[b * 2 + 0] = v[0] * ow_[0] + ob[0];
  out[b * 2 + 1] = v[1] * ow_[1] + ob[1];
}

// -------------------------------------------------------------------------
extern "C" void kernel_launch(void* const* d_in, const int* in_sizes, int n_in,
                              void* d_out, int out_size, void* d_ws, size_t ws_size,
                              hipStream_t stream)
{
  const float* x     = (const float*)d_in[0];
  const float* w1    = (const float*)d_in[1];
  const float* b1    = (const float*)d_in[2];
  const float* w2    = (const float*)d_in[3];
  const float* b2    = (const float*)d_in[4];
  const float* gleak = (const float*)d_in[5];
  const float* vleak = (const float*)d_in[6];
  const float* cm    = (const float*)d_in[7];
  const float* sigma = (const float*)d_in[8];
  const float* mu    = (const float*)d_in[9];
  const float* w     = (const float*)d_in[10];
  const float* erev  = (const float*)d_in[11];
  const float* ssig  = (const float*)d_in[12];
  const float* smu   = (const float*)d_in[13];
  const float* sw    = (const float*)d_in[14];
  const float* serev = (const float*)d_in[15];
  const float* iw    = (const float*)d_in[16];
  const float* ib    = (const float*)d_in[17];
  const float* ow_   = (const float*)d_in[18];
  const float* ob    = (const float*)d_in[19];
  const float* smask = (const float*)d_in[20];
  const float* ssmask= (const float*)d_in[21];

  // workspace layout: pooled1 f16 (B*6*12*20 = 11,796,480 elems = 23,592,960 B),
  // then feat f32 (B*512 = 16,777,216 B). Total ~40.4 MB.
  _Float16* pooled1 = (_Float16*)d_ws;
  float*    feat    = (float*)((char*)d_ws + 23592960);

  conv1_pool_kernel<<<BATCH, 256, 0, stream>>>(x, w1, b1, pooled1);
  conv2_wmma_kernel<<<BATCH, 256, 0, stream>>>(pooled1, w2, b2, feat);
  ltc_kernel<<<32, 256, 0, stream>>>(feat, gleak, vleak, cm, sigma, mu, w, erev,
                                     ssig, smu, sw, serev, iw, ib, ow_, ob,
                                     smask, ssmask, (float*)d_out);
}
